// HeatmapHead_14577119003128
// MI455X (gfx1250) — matrix operand order repaired and checked by reading, not compiled, not run on hardware
//
#include <hip/hip_runtime.h>
#include <hip/hip_bf16.h>

// ---------------------------------------------------------------------------
// Siamese tracking head for MI455X (gfx1250, wave32, WMMA).
//
// Stage 1: depthwise xcorr (VALU FMA from LDS tiles); writes corr TRANSPOSED
//          as [n][25*25 spatial][256 channel] bf16 so stage 2's im2col reads
//          are contiguous along the WMMA K axis. 2.0 GFLOP, memory-bound
//          (126 MB x_f in, 39 MB bf16 corr out -> ~7 us HBM floor).
// Stage 2: heat+reg 3x3 convs fused as one GEMM per image:
//          M = 529 spatial, N = 5 outputs (padded to 16),
//          K = 2304 ordered t = tap*256 + channel, so each 32-wide K-chunk
//          has a wave-uniform tap and a contiguous channel run:
//            A fragment = 2x global_load_b128 per lane,
//            B fragment = 2x ds_load_b128 per lane (weights staged in LDS),
//          72 x v_wmma_f32_16x16x32_bf16 per M-tile (8 unrolled per tap).
// corr (39 MB bf16) stays resident in the 192 MB L2 for stage-2 re-reads.
// ---------------------------------------------------------------------------

typedef __attribute__((ext_vector_type(16))) __bf16 v16bf;
typedef __attribute__((ext_vector_type(8)))  __bf16 v8bf;
typedef __attribute__((ext_vector_type(8)))  float  v8f;

#define N_IMG 128
#define CHN   256
#define HF 31
#define WF 31
#define XSP (HF*WF)      // 961
#define HK 7
#define WK 7
#define ZSP (HK*WK)      // 49
#define HO 25
#define WO 25
#define CORR_SP (HO*WO)  // 625
#define HP 23
#define WP 23
#define OUT_SP (HP*WP)   // 529
#define KTOT  (CHN*9)    // 2304
#define MTILES ((OUT_SP + 15) / 16)  // 34
#define HEAT_ELEMS (N_IMG*OUT_SP)    // 67712
#define CGRP 16                       // channels per stage-1 block

// ------------------------------ Stage 1 ------------------------------------
// grid.x = N_IMG * (CHN/CGRP) = 2048 blocks, 256 threads.
// Block stages 16 channels of x_f (float4-coalesced) + templates in LDS;
// thread (c_local = tid&15) computes its channel at spatial stride 16 and
// stores bf16 into the channel-minor transposed corr buffer (coalesced).
__global__ __launch_bounds__(256)
void xcorr_kernel(const float* __restrict__ xf,
                  const float* __restrict__ zf,
                  __bf16* __restrict__ corr)   // [n][625][256]
{
    __shared__ __align__(16) float xs[CGRP * XSP];   // 61.5 KB
    __shared__ __align__(16) float zs[CGRP * ZSP];   // 3.1 KB

    const int n   = blockIdx.x >> 4;
    const int cg  = blockIdx.x & 15;
    const int c0  = cg * CGRP;
    const int tid = threadIdx.x;

    const float* xin = xf + (size_t)(n * CHN + c0) * XSP;  // 16B-aligned base
    const float* zin = zf + (size_t)(n * CHN + c0) * ZSP;  // 16B-aligned base

    // float4 staging: 16*961 = 15376 floats (exactly 3844 float4s)
    for (int i = tid; i < (CGRP * XSP) / 4; i += 256)
        ((float4*)xs)[i] = ((const float4*)xin)[i];
    for (int i = tid; i < (CGRP * ZSP) / 4; i += 256)
        ((float4*)zs)[i] = ((const float4*)zin)[i];
    __syncthreads();

    const int c_local = tid & 15;
    const float* xrow = xs + c_local * XSP;
    const float* zrow = zs + c_local * ZSP;
    __bf16* cout = corr + (size_t)n * CORR_SP * CHN + (c0 + c_local);

    for (int idx = (tid >> 4); idx < CORR_SP; idx += 16) {
        const int p = idx / WO;
        const int q = idx - p * WO;
        float acc = 0.0f;
#pragma unroll
        for (int i = 0; i < HK; ++i) {
#pragma unroll
            for (int j = 0; j < WK; ++j) {
                acc = __builtin_fmaf(xrow[(p + i) * WF + (q + j)],
                                     zrow[i * WK + j], acc);
            }
        }
        cout[(size_t)idx * CHN] = (__bf16)acc;
    }
}

// ------------------------------ Stage 2 ------------------------------------
// grid = (N_IMG, 5), 256 threads = 8 waves; wave owns one 16-row M-tile.
__global__ __launch_bounds__(256)
void head_kernel(const __bf16* __restrict__ corr,   // [n][625][256]
                 const float* __restrict__ heat_w,
                 const float* __restrict__ heat_b,
                 const float* __restrict__ reg_w,
                 const float* __restrict__ reg_b,
                 float* __restrict__ out)
{
    // Weights row-major per output channel: wlds[o][t2], t2 = tap*256 + c.
    // Rows 5..7 are zero; lanes l16 >= 8 alias the zero row.
    __shared__ __align__(16) __bf16 wlds[8 * KTOT];  // 36 KB

    const int n   = blockIdx.x;
    const int tid = threadIdx.x;

#pragma unroll
    for (int o = 0; o < 8; ++o) {
        for (int i = tid; i < KTOT; i += 256) {
            const int r = i >> 8;       // tap 0..8
            const int c = i & 255;      // channel
            __bf16 v = (__bf16)0.0f;
            if (o == 0)      v = (__bf16)heat_w[c * 9 + r];
            else if (o < 5)  v = (__bf16)reg_w[(o - 1) * KTOT + c * 9 + r];
            wlds[o * KTOT + i] = v;
        }
    }
    __syncthreads();

    const int wave = tid >> 5;
    const int lane = tid & 31;
    const int h    = lane >> 4;     // 16-lane half
    const int l16  = lane & 15;

    const int tile = blockIdx.y * 8 + wave;
    if (tile >= MTILES) return;     // wave-uniform; EXEC stays full elsewhere

    // This lane's im2col M row. Rows m >= 529 use a clamped address; their
    // garbage only feeds D rows we never store.
    const int mA  = tile * 16 + l16;
    const int mcl = (mA < OUT_SP) ? mA : (OUT_SP - 1);
    const int ya  = mcl / WP;
    const int xa  = mcl - ya * WP;

    const __bf16* corr_n = corr + (size_t)n * CORR_SP * CHN;
    const __bf16* brow   = wlds + (size_t)((l16 < 8) ? l16 : 7) * KTOT + 8 * h;

    v8f acc = {};
    for (int r9 = 0; r9 < 9; ++r9) {              // 3x3 tap (wave-uniform)
        const int dy = r9 / 3;
        const int dx = r9 - dy * 3;
        const __bf16* arow =
            corr_n + (size_t)((ya + dy) * WO + (xa + dx)) * CHN + 8 * h;
        const __bf16* bb = brow + r9 * 256;       // LDS, contiguous K run

        // L2 prefetch next tap's gather row.
        __builtin_prefetch(arow + WO * CHN, 0, 0);

#pragma unroll
        for (int kc = 0; kc < 8; ++kc) {          // 8 channel chunks of 32
            // A fragment: e<8 -> K=8h+e ; e>=8 -> K=16+8h+(e-8)
            const v8bf alo = *(const v8bf*)(arow + kc * 32);
            const v8bf ahi = *(const v8bf*)(arow + kc * 32 + 16);
            // B fragment: e<8 -> K=8h+e ; e>=8 -> K=16+8h+(e-8)
            const v8bf blo = *(const v8bf*)(bb + kc * 32);
            const v8bf bhi = *(const v8bf*)(bb + kc * 32 + 16);

            const v16bf a = __builtin_shufflevector(
                alo, ahi, 0, 1, 2, 3, 4, 5, 6, 7, 8, 9, 10, 11, 12, 13, 14, 15);
            const v16bf b = __builtin_shufflevector(
                blo, bhi, 0, 1, 2, 3, 4, 5, 6, 7, 8, 9, 10, 11, 12, 13, 14, 15);

            acc = __builtin_amdgcn_wmma_f32_16x16x32_bf16(
                      /*neg_a=*/false, a, /*neg_b=*/false, b,
                      /*c_mod=*/(short)0, acc,
                      /*reuse_a=*/false, /*reuse_b=*/false);
        }
    }

    // D layout: VGPR r holds M = r + 8h, N = lane%16.
    const int o = l16;
    if (o < 5) {
        const float bias = (o == 0) ? heat_b[0] : reg_b[o - 1];
#pragma unroll
        for (int r = 0; r < 8; ++r) {
            const int m = tile * 16 + r + 8 * h;
            if (m < OUT_SP) {
                const float v = acc[r] + bias;
                if (o == 0) {
                    out[(size_t)n * OUT_SP + m] = v;
                } else {
                    out[(size_t)HEAT_ELEMS
                        + ((size_t)(n * 4 + (o - 1))) * OUT_SP + m] = v;
                }
            }
        }
    }
}

// ---------------------------------------------------------------------------
extern "C" void kernel_launch(void* const* d_in, const int* in_sizes, int n_in,
                              void* d_out, int out_size, void* d_ws, size_t ws_size,
                              hipStream_t stream)
{
    const float* xf = (const float*)d_in[0];   // (128,256,31,31)
    const float* zf = (const float*)d_in[1];   // (128,256,7,7)
    const float* hw = (const float*)d_in[2];   // (1,256,3,3)
    const float* hb = (const float*)d_in[3];   // (1,)
    const float* rw = (const float*)d_in[4];   // (4,256,3,3)
    const float* rb = (const float*)d_in[5];   // (4,)

    // corr intermediate, channel-minor bf16: 128*625*256*2 B = ~39 MB in d_ws.
    __bf16* corr = (__bf16*)d_ws;
    float*  out  = (float*)d_out;

    xcorr_kernel<<<dim3(N_IMG * (CHN / CGRP)), dim3(256), 0, stream>>>(
        xf, zf, corr);
    head_kernel<<<dim3(N_IMG, (MTILES + 7) / 8), dim3(256), 0, stream>>>(
        corr, hw, hb, rw, rb, out);
}
